// OrthoLinear_3865470566791
// MI455X (gfx1250) — matrix operand order repaired and checked by reading, not compiled, not run on hardware
//
#include <hip/hip_runtime.h>
#include <hip/hip_bf16.h>
#include <hip/hip_fp16.h>

typedef __attribute__((ext_vector_type(16))) _Float16 v16h;
typedef __attribute__((ext_vector_type(8)))  float    v8f;

#define OUTF  4096   // output features (rows of W)
#define INF   4096   // input features  (K)
#define KPACK 2048   // packed bytes per row

// ---- CDNA5 async global->LDS support (guarded; falls back to sync copy) ----
#if defined(__has_builtin)
#if __has_builtin(__builtin_amdgcn_global_load_async_to_lds_b128)
#define HAVE_ASYNC_LDS 1
#endif
#if __has_builtin(__builtin_amdgcn_s_wait_asynccnt)
#define HAVE_ASYNC_WAIT_BUILTIN 1
#endif
#endif

// Builtin (ROCm 7.2 clang-22) expects: (int-vec16B __device__*, ... , Ii, Ii)
typedef int v4i32 __attribute__((vector_size(4 * sizeof(int))));
typedef __attribute__((address_space(1))) v4i32* gv4p;   // prints as __device__*
typedef __attribute__((address_space(3))) v4i32* lv4p;   // prints as __shared__*

__device__ __forceinline__ void copy16_async(const _Float16* g, _Float16* l) {
#if defined(HAVE_ASYNC_LDS)
  __builtin_amdgcn_global_load_async_to_lds_b128((gv4p)g, (lv4p)l, 0, 0);
#else
  *(uint4*)l = *(const uint4*)g;
#endif
}

__device__ __forceinline__ void wait_async_group() {
#if defined(HAVE_ASYNC_LDS)
#if defined(HAVE_ASYNC_WAIT_BUILTIN)
  __builtin_amdgcn_s_wait_asynccnt(0);
#else
  asm volatile("s_wait_asynccnt 0x0" ::: "memory");
#endif
#endif
}

// ---------------------------------------------------------------------------
// Pass 1: dequantize int4 (offset-binary) weights into f16 W[o][k] in ws.
// ---------------------------------------------------------------------------
__global__ __launch_bounds__(256) void dequant_kernel(
    const int* __restrict__ packed, const float* __restrict__ scales,
    _Float16* __restrict__ W) {
  int idx = blockIdx.x * 256 + threadIdx.x;       // 0 .. OUTF*KPACK-1
  int o = idx >> 11;                              // / KPACK
  int j = idx & (KPACK - 1);
  int p = packed[idx];
  float s = scales[o];
  float lo = (float)((p & 0xF) - 8) * s;
  float hi = (float)(((p >> 4) & 0xF) - 8) * s;
  union { _Float16 h[2]; unsigned int u; } pk;
  pk.h[0] = (_Float16)lo;
  pk.h[1] = (_Float16)hi;
  *(unsigned int*)(&W[(size_t)o * INF + 2 * j]) = pk.u;
}

// ---------------------------------------------------------------------------
// Pass 2: scatter-add sparse outliers via packed-f16 global atomic (dev scope).
// ---------------------------------------------------------------------------
__global__ __launch_bounds__(256) void scatter_kernel(
    const float* __restrict__ vals, const int* __restrict__ rows,
    const int* __restrict__ cols, _Float16* __restrict__ W, int nnz) {
  int idx = blockIdx.x * 256 + threadIdx.x;
  if (idx >= nnz) return;
  int r = rows[idx];
  int c = cols[idx];
  _Float16 hv = (_Float16)vals[idx];
  union { _Float16 h[2]; unsigned int u; } pk;
  pk.h[0] = (c & 1) ? (_Float16)0.0f : hv;
  pk.h[1] = (c & 1) ? hv : (_Float16)0.0f;
  _Float16* addr = W + (size_t)r * INF + (c & ~1);
  asm volatile("global_atomic_pk_add_f16 %0, %1, off scope:SCOPE_DEV"
               :: "v"(addr), "v"(pk.u) : "memory");
}

// ---------------------------------------------------------------------------
// Pass 2b: x fp32 -> f16 (one-time; keeps cvt out of the GEMM hot loop).
// ---------------------------------------------------------------------------
__global__ __launch_bounds__(256) void cvt_x_kernel(
    const float* __restrict__ X, _Float16* __restrict__ Xh) {
  size_t idx = (size_t)blockIdx.x * 256 + threadIdx.x;   // one per 4 floats
  float4 f = *(const float4*)(X + idx * 4);
  union { _Float16 h[4]; unsigned long long u; } pk;
  pk.h[0] = (_Float16)f.x; pk.h[1] = (_Float16)f.y;
  pk.h[2] = (_Float16)f.z; pk.h[3] = (_Float16)f.w;
  *(unsigned long long*)(Xh + idx * 4) = pk.u;
}

// ---------------------------------------------------------------------------
// Pass 3 (fast path): GEMM out[m][o] = sum_k Xh[m][k] * W[o][k]
// 128x128 block tile, BK=32, 8 waves (2x4), wave tile 64x32 = 4x2 WMMAs.
// Double-buffered LDS fed by async global->LDS b128; 1 barrier / K-step.
// ---------------------------------------------------------------------------
__global__ __launch_bounds__(256) void gemm_f16_kernel(
    const _Float16* __restrict__ Xh, const _Float16* __restrict__ W,
    float* __restrict__ out) {
  __shared__ _Float16 Ah[2][128][40];   // [buf][m][k], 80B row stride
  __shared__ _Float16 Bh[2][128][40];   // [buf][n][k]

  const int tid  = threadIdx.x;
  const int lane = tid & 31;
  const int wave = tid >> 5;
  const int wm   = wave >> 2;          // 0..1 -> 64-row slice
  const int wn   = wave & 3;           // 0..3 -> 32-col slice
  const int nl   = lane & 15;
  const int kh   = lane >> 4;
  const int m0   = blockIdx.y * 128;
  const int n0   = blockIdx.x * 128;

  // Per-thread staging assignment: row rr (0..127), 16-half chunk `part`.
  const int rr   = tid >> 1;
  const int part = (tid & 1) * 16;
  const _Float16* gA = Xh + (size_t)(m0 + rr) * INF + part;
  const _Float16* gB = W  + (size_t)(n0 + rr) * INF + part;

  v8f acc[4][2];
#pragma unroll
  for (int i = 0; i < 4; ++i)
#pragma unroll
    for (int j = 0; j < 2; ++j) acc[i][j] = {};

  // Prologue: stage tile 0 into buffer 0.
  copy16_async(gA, &Ah[0][rr][part]);
  copy16_async(gB, &Bh[0][rr][part]);

  const int T = INF / 32;
  for (int t = 0; t < T; ++t) {
    const int buf = t & 1;
    wait_async_group();       // own async group for tile t has landed in LDS
    __syncthreads();          // everyone's group landed; prev reads of buf done

    if (t + 1 < T) {          // stage tile t+1 into the other buffer
      if (t + 2 < T)
        __builtin_prefetch(gB + (size_t)(t + 2) * 32, 0, 1);
      copy16_async(gA + (size_t)(t + 1) * 32, &Ah[buf ^ 1][rr][part]);
      copy16_async(gB + (size_t)(t + 1) * 32, &Bh[buf ^ 1][rr][part]);
    }

    // Fragments per ISA wave32 layouts.
    v16h afrag[4], bfrag[2];
#pragma unroll
    for (int i = 0; i < 4; ++i) {
      int row = wm * 64 + i * 16 + nl;
      int kb  = kh * 8;
      union { uint4 u[2]; v16h v; } fr;
      fr.u[0] = *(const uint4*)&Ah[buf][row][kb];
      fr.u[1] = *(const uint4*)&Ah[buf][row][kb + 16];
      afrag[i] = fr.v;
    }
#pragma unroll
    for (int j = 0; j < 2; ++j) {
      int col = wn * 32 + j * 16 + nl;
      int kb  = kh * 8;
      union { uint4 u[2]; v16h v; } fr;
      fr.u[0] = *(const uint4*)&Bh[buf][col][kb];
      fr.u[1] = *(const uint4*)&Bh[buf][col][kb + 16];
      bfrag[j] = fr.v;
    }
#pragma unroll
    for (int i = 0; i < 4; ++i)
#pragma unroll
      for (int j = 0; j < 2; ++j)
        acc[i][j] = __builtin_amdgcn_wmma_f32_16x16x32_f16(
            false, afrag[i], false, bfrag[j], (short)0, acc[i][j],
            false, false);
  }

  // Epilogue: C layout = VGPR r -> M = (lane>=16)*8 + r, N = lane&15.
#pragma unroll
  for (int i = 0; i < 4; ++i)
#pragma unroll
    for (int j = 0; j < 2; ++j) {
      int oc = n0 + wn * 32 + j * 16 + nl;
      int mb = m0 + wm * 64 + i * 16 + kh * 8;
#pragma unroll
      for (int r = 0; r < 8; ++r)
        out[(size_t)(mb + r) * OUTF + oc] = acc[i][j][r];
    }
}

// ---------------------------------------------------------------------------
// Pass 3 (fallback if ws too small for Xh): converts x fp32->f16 in-kernel.
// ---------------------------------------------------------------------------
__global__ __launch_bounds__(256) void gemm_f32x_kernel(
    const float* __restrict__ X, const _Float16* __restrict__ W,
    float* __restrict__ out) {
  __shared__ _Float16 Ah[128][40];
  __shared__ _Float16 Bh[128][40];

  const int tid  = threadIdx.x;
  const int lane = tid & 31;
  const int wave = tid >> 5;
  const int wm   = wave >> 2;
  const int wn   = wave & 3;
  const int nl   = lane & 15;
  const int kh   = lane >> 4;
  const int m0   = blockIdx.y * 128;
  const int n0   = blockIdx.x * 128;

  v8f acc[4][2];
#pragma unroll
  for (int i = 0; i < 4; ++i)
#pragma unroll
    for (int j = 0; j < 2; ++j) acc[i][j] = {};

  for (int k0 = 0; k0 < INF; k0 += 32) {
    __syncthreads();
    {
      int r  = tid >> 3;
      int kc = (tid & 7) * 4;
#pragma unroll
      for (int q = 0; q < 4; ++q) {
        int row = r + q * 32;
        float4 f = *(const float4*)(X + (size_t)(m0 + row) * INF + k0 + kc);
        union { _Float16 h[4]; unsigned long long u; } pk;
        pk.h[0] = (_Float16)f.x; pk.h[1] = (_Float16)f.y;
        pk.h[2] = (_Float16)f.z; pk.h[3] = (_Float16)f.w;
        *(unsigned long long*)&Ah[row][kc] = pk.u;
      }
    }
    {
      int n    = tid >> 1;
      int part = (tid & 1) * 16;
      const uint4* src = (const uint4*)(W + (size_t)(n0 + n) * INF + k0 + part);
      uint4 b0 = src[0];
      uint4 b1 = src[1];
      *(uint4*)&Bh[n][part]     = b0;
      *(uint4*)&Bh[n][part + 8] = b1;
    }
    __syncthreads();

    v16h afrag[4], bfrag[2];
#pragma unroll
    for (int i = 0; i < 4; ++i) {
      int row = wm * 64 + i * 16 + nl;
      int kb  = kh * 8;
      union { uint4 u[2]; v16h v; } fr;
      fr.u[0] = *(const uint4*)&Ah[row][kb];
      fr.u[1] = *(const uint4*)&Ah[row][kb + 16];
      afrag[i] = fr.v;
    }
#pragma unroll
    for (int j = 0; j < 2; ++j) {
      int col = wn * 32 + j * 16 + nl;
      int kb  = kh * 8;
      union { uint4 u[2]; v16h v; } fr;
      fr.u[0] = *(const uint4*)&Bh[col][kb];
      fr.u[1] = *(const uint4*)&Bh[col][kb + 16];
      bfrag[j] = fr.v;
    }
#pragma unroll
    for (int i = 0; i < 4; ++i)
#pragma unroll
      for (int j = 0; j < 2; ++j)
        acc[i][j] = __builtin_amdgcn_wmma_f32_16x16x32_f16(
            false, afrag[i], false, bfrag[j], (short)0, acc[i][j],
            false, false);
  }

#pragma unroll
  for (int i = 0; i < 4; ++i)
#pragma unroll
    for (int j = 0; j < 2; ++j) {
      int oc = n0 + wn * 32 + j * 16 + nl;
      int mb = m0 + wm * 64 + i * 16 + kh * 8;
#pragma unroll
      for (int r = 0; r < 8; ++r)
        out[(size_t)(mb + r) * OUTF + oc] = acc[i][j][r];
    }
}

// ---------------------------------------------------------------------------
extern "C" void kernel_launch(void* const* d_in, const int* in_sizes, int n_in,
                              void* d_out, int out_size, void* d_ws, size_t ws_size,
                              hipStream_t stream) {
  const float* x      = (const float*)d_in[0];
  const int*   packed = (const int*)d_in[1];
  const float* scales = (const float*)d_in[2];
  const float* vals   = (const float*)d_in[3];
  const int*   rows   = (const int*)d_in[4];
  const int*   cols   = (const int*)d_in[5];
  float*       out    = (float*)d_out;

  const int nnz = in_sizes[3];
  const int M   = in_sizes[0] / INF;               // 4096 tokens

  const size_t needW = (size_t)OUTF * INF * sizeof(_Float16);  // 32 MB
  const size_t needX = (size_t)M * INF * sizeof(_Float16);     // 32 MB

  _Float16* W = (_Float16*)d_ws;

  dequant_kernel<<<(OUTF * KPACK) / 256, 256, 0, stream>>>(packed, scales, W);
  scatter_kernel<<<(nnz + 255) / 256, 256, 0, stream>>>(vals, rows, cols, W, nnz);

  if (ws_size >= needW + needX) {
    _Float16* Xh = (_Float16*)((char*)d_ws + needW);
    cvt_x_kernel<<<((size_t)M * INF / 4 + 255) / 256, 256, 0, stream>>>(x, Xh);
    gemm_f16_kernel<<<dim3(OUTF / 128, M / 128), 256, 0, stream>>>(Xh, W, out);
  } else {
    gemm_f32x_kernel<<<dim3(OUTF / 128, M / 128), 256, 0, stream>>>(x, W, out);
  }
}
